// NeuralODE_46153718563178
// MI455X (gfx1250) — matrix operand order repaired and checked by reading, not compiled, not run on hardware
//
#include <hip/hip_runtime.h>
#include <hip/hip_bf16.h>
#include <math.h>

typedef __attribute__((ext_vector_type(16))) __bf16 v16bf;
typedef __attribute__((ext_vector_type(8)))  float  v8f;

#define B_   1024
#define D_   128
#define W_   512
#define T_   8
#define BD   131072   /* B_*D_ */
#define BW   524288   /* B_*W_ */
#define NSTEP 48

/* float scalar slots */
#define SC_T    0
#define SC_DT   1
#define SC_H    2
#define SC_TNS  3
#define SC_ACC  4
/* int scalar slots (at sc+8 floats) */
#define SI_SI   0
#define SI_DONE 1
#define SI_ACCE 2
#define SI_HIT  3
#define SI_SAVE 4

/* Tsit5 tableau (f32, matching jnp.float32 math) */
static constexpr float A21c = 0.161f;
static constexpr float A31c = -0.008480655492356989f, A32c = 0.335480655492357f;
static constexpr float A41c = 2.8971530571054935f, A42c = -6.359448489975075f, A43c = 4.3622954328695815f;
static constexpr float A51c = 5.325864828439257f,  A52c = -11.748883564062828f, A53c = 7.4955393428898365f, A54c = -0.09249506636175525f;
static constexpr float A61c = 5.86145544294642f,   A62c = -12.92096931784711f,  A63c = 8.159367898576159f,  A64c = -0.071584973281401f, A65c = -0.028269050394068383f;
static constexpr float B1c = 0.09646076681806523f, B2c = 0.01f, B3c = 0.4798896504144996f, B4c = 1.379008574103742f, B5c = -3.290069515436081f, B6c = 2.324710524099774f;
static constexpr float E1c = -0.00178001105222577714f, E2c = -0.0008164344596567469f, E3c = 0.007880878010261995f;
static constexpr float E4c = -0.1447110071732629f, E5c = 0.5823571654525552f, E6c = -0.45808210592918697f, E7c = 0.015151515151515152f;
static constexpr float RTOLc = 1e-3f, ATOLc = 1e-6f;
static constexpr float SAFETYc = 0.9f, FMINc = 0.2f, FMAXc = 10.0f;

/* ---------------- weight split: f32 -> bf16 hi + bf16 lo ---------------- */
__global__ void split_bf16_k(const float* __restrict__ w, __bf16* __restrict__ hi,
                             __bf16* __restrict__ lo, int n)
{
    int i = blockIdx.x * blockDim.x + threadIdx.x;
    if (i >= n) return;
    float v = w[i];
    __bf16 h = (__bf16)v;
    hi[i] = h;
    lo[i] = (__bf16)(v - (float)h);
}

/* ---------------- WMMA GEMM: C[M,N] = act(A[M,K] @ W^T + bias) ----------------
 * W stored [N,K] row-major; pre-split into bf16 hi/lo.  A split on the fly.
 * One wave -> 16 x (16*NBLK) D tile: one shared A fragment, NBLK accumulators,
 * 3*NBLK WMMAs per K=32 chunk (split-precision AhBh + AlBh + AhBl).
 * All NBLK hi/lo B fragments preloaded into distinct registers per chunk so
 * load latency hides under the A-split VALU work.  K,N compile-time constants.
 * Block = 8 waves stacked in M (128 rows). D layout per CDNA5 ISA 7.12.2; A/B use contiguous k = 16*lh + e. */
template<int NBLK, int K, int N, bool ACT>
__global__ __launch_bounds__(256)
void wmma_gemm_bias(const float* __restrict__ A, const __bf16* __restrict__ Bhi,
                    const __bf16* __restrict__ Blo, const float* __restrict__ bias,
                    float* __restrict__ C)
{
    const int wave = threadIdx.x >> 5;
    const int lane = threadIdx.x & 31;
    const int l16  = lane & 15;
    const int lh   = lane >> 4;              /* lane-half: selects contiguous K half (k = 16*lh + e) */
    const int N0 = blockIdx.x * (16 * NBLK);
    const int M0 = (blockIdx.y << 7) + (wave << 4);

    const float* Arow = A + (M0 + l16) * K;
    const __bf16* bhp = Bhi + (N0 + l16) * K + (lh << 4);
    const __bf16* blp = Blo + (N0 + l16) * K + (lh << 4);
    const int aoff = lh << 4;

    v8f c[NBLK];
#pragma unroll
    for (int j = 0; j < NBLK; ++j) c[j] = {};

    for (int kc = 0; kc < K; kc += 32) {
        /* ---- issue all B-fragment loads first (distinct registers) ---- */
        v16bf bh[NBLK], bl[NBLK];
#pragma unroll
        for (int j = 0; j < NBLK; ++j) {
            bh[j] = *(const v16bf*)(bhp + j * 16 * K + kc);
            bl[j] = *(const v16bf*)(blp + j * 16 * K + kc);
        }
        /* ---- A fragment (16x32 bf16 layout) + split, overlaps load latency ---- */
        float af[16];
#pragma unroll
        for (int e = 0; e < 8; ++e) af[e]     = Arow[kc + aoff + e];
#pragma unroll
        for (int e = 0; e < 8; ++e) af[8 + e] = Arow[kc + aoff + 8 + e];
        v16bf ahi, alo;
#pragma unroll
        for (int e = 0; e < 16; ++e) {
            __bf16 h = (__bf16)af[e];
            ahi[e] = h;
            alo[e] = (__bf16)(af[e] - (float)h);
        }
        if (kc + 32 < K) {                     /* gfx1250 global_prefetch_b8 */
            __builtin_prefetch(Arow + kc + 32, 0, 1);
            __builtin_prefetch(bhp + kc + 32, 0, 1);
        }
        /* ---- 3*NBLK WMMAs: NBLK independent accumulator chains ---- */
#pragma unroll
        for (int j = 0; j < NBLK; ++j) {
            c[j] = __builtin_amdgcn_wmma_f32_16x16x32_bf16(false, ahi, false, bh[j], (short)0, c[j], false, false);
            c[j] = __builtin_amdgcn_wmma_f32_16x16x32_bf16(false, alo, false, bh[j], (short)0, c[j], false, false);
            c[j] = __builtin_amdgcn_wmma_f32_16x16x32_bf16(false, ahi, false, bl[j], (short)0, c[j], false, false);
        }
    }
#pragma unroll
    for (int j = 0; j < NBLK; ++j) {
        const int col = N0 + j * 16 + l16;
        const float bv = bias[col];
#pragma unroll
        for (int r = 0; r < 8; ++r) {          /* D: vgpr r -> row M0 + lh*8 + r */
            float v = c[j][r] + bv;
            if (ACT) v = fmaxf(v, 0.0f) + log1pf(__expf(-fabsf(v)));  /* softplus */
            C[(M0 + (lh << 3) + r) * N + col] = v;
        }
    }
}

/* ---------------- RK stage combine: out = y + h * sum(ci*ki) ---------------- */
__global__ __launch_bounds__(256)
void combine_k(float* __restrict__ out, const float* __restrict__ yb,
               const float* __restrict__ sc,
               const float* __restrict__ k1, const float* __restrict__ k2,
               const float* __restrict__ k3, const float* __restrict__ k4,
               const float* __restrict__ k5, const float* __restrict__ k6,
               float c1, float c2, float c3, float c4, float c5, float c6, int count)
{
    int i = blockIdx.x * blockDim.x + threadIdx.x;
    if (i >= BD) return;
    float h = sc[SC_H];
    float s = c1 * k1[i];
    if (count > 1) s += c2 * k2[i];
    if (count > 2) s += c3 * k3[i];
    if (count > 3) s += c4 * k4[i];
    if (count > 4) s += c5 * k5[i];
    if (count > 5) s += c6 * k6[i];
    out[i] = yb[i] + h * s;
}

/* ---------------- error norm: sum((err/scale)^2) -> sc[SC_ACC] ---------------- */
__global__ __launch_bounds__(256)
void err_reduce_k(const float* __restrict__ k1, const float* __restrict__ k2,
                  const float* __restrict__ k3, const float* __restrict__ k4,
                  const float* __restrict__ k5, const float* __restrict__ k6,
                  const float* __restrict__ k7,
                  const float* __restrict__ y, const float* __restrict__ ynew,
                  float* sc)
{
    float h = sc[SC_H];
    float local = 0.0f;
    for (int i = blockIdx.x * blockDim.x + threadIdx.x; i < BD; i += gridDim.x * blockDim.x) {
        float e = h * (E1c*k1[i] + E2c*k2[i] + E3c*k3[i] + E4c*k4[i]
                     + E5c*k5[i] + E6c*k6[i] + E7c*k7[i]);
        float s = ATOLc + RTOLc * fmaxf(fabsf(y[i]), fabsf(ynew[i]));
        float r = e / s;
        local += r * r;
    }
#pragma unroll
    for (int o = 16; o > 0; o >>= 1) local += __shfl_down(local, o, 32);
    __shared__ float red[8];
    if ((threadIdx.x & 31) == 0) red[threadIdx.x >> 5] = local;
    __syncthreads();
    if (threadIdx.x == 0) {
        float s = 0.0f;
#pragma unroll
        for (int w = 0; w < 8; ++w) s += red[w];
        atomicAdd(&sc[SC_ACC], s);
    }
}

/* ---------------- per-step scalar control ---------------- */
__global__ void step_begin_k(const float* __restrict__ ts, float* sc, int* sci)
{
    int si = sci[SI_SI];
    int done = (si >= T_) ? 1 : 0;
    int si_c = (si < T_ - 1) ? si : (T_ - 1);
    float tns = ts[si_c];
    float h = done ? 0.0f : fmaxf(fminf(sc[SC_DT], tns - sc[SC_T]), 0.0f);
    sc[SC_H] = h; sc[SC_TNS] = tns; sc[SC_ACC] = 0.0f;
    sci[SI_DONE] = done; sci[SI_SAVE] = si_c;
}

__global__ void step_end_k(float* sc, int* sci)
{
    float enorm = sqrtf(sc[SC_ACC] / (float)BD);
    int done   = sci[SI_DONE];
    int accept = (enorm <= 1.0f) && !done;
    float tn   = sc[SC_T] + sc[SC_H];
    int hit    = accept && (tn >= sc[SC_TNS] - 1e-6f);
    if (accept) sc[SC_T] = tn;
    sci[SI_ACCE] = accept; sci[SI_HIT] = hit;
    sci[SI_SI] += hit;
    if (!done) {
        float factor = FMAXc;
        if (enorm > 0.0f)
            factor = fminf(fmaxf(SAFETYc * __powf(enorm, -0.2f), FMINc), FMAXc);
        sc[SC_DT] *= factor;
    }
}

__global__ __launch_bounds__(256)
void update_k(float* __restrict__ y, const float* __restrict__ ynew,
              float* __restrict__ ys, const int* sci)
{
    int i = blockIdx.x * blockDim.x + threadIdx.x;
    if (i >= BD) return;
    float v = sci[SI_ACCE] ? ynew[i] : y[i];
    y[i] = v;
    if (sci[SI_HIT]) ys[sci[SI_SAVE] * BD + i] = v;
}

__global__ __launch_bounds__(256)
void init_k(const float* __restrict__ ts, const float* __restrict__ y0,
            float* __restrict__ y, float* __restrict__ ys, float* sc, int* sci)
{
    int i = blockIdx.x * blockDim.x + threadIdx.x;
    if (i < T_ * BD) ys[i] = (i < BD) ? y0[i] : 0.0f;   /* SaveAt includes t0 */
    if (i < BD) y[i] = y0[i];
    if (i == 0) {
        sc[SC_T]  = ts[0];
        sc[SC_DT] = ts[1] - ts[0];
        sci[SI_SI] = 1;
    }
}

/* ---------------- host driver (graph-capturable, deterministic) ---------------- */
extern "C" void kernel_launch(void* const* d_in, const int* in_sizes, int n_in,
                              void* d_out, int out_size, void* d_ws, size_t ws_size,
                              hipStream_t stream)
{
    (void)in_sizes; (void)n_in; (void)out_size; (void)ws_size;
    const float* ts = (const float*)d_in[0];
    const float* y0 = (const float*)d_in[1];
    const float* W0 = (const float*)d_in[2];
    const float* b0 = (const float*)d_in[3];
    const float* W1 = (const float*)d_in[4];
    const float* b1 = (const float*)d_in[5];
    const float* W2 = (const float*)d_in[6];
    const float* b2 = (const float*)d_in[7];
    float* ys = (float*)d_out;
    float* f  = (float*)d_ws;

    float* y    = f;
    float* ynew = f + BD;
    float* ytmp = f + 2 * BD;
    float* k[7];
    for (int i = 0; i < 7; ++i) k[i] = f + (3 + i) * BD;
    float* h1 = f + 10 * BD;
    float* h2 = h1 + BW;
    float* sc = h2 + BW;
    int*   sci = (int*)(sc + 8);
    __bf16* wb   = (__bf16*)(sc + 32);      /* 32B-aligned bf16 weight region */
    __bf16* W0hi = wb;               __bf16* W0lo = W0hi + W_ * D_;
    __bf16* W1hi = W0lo + W_ * D_;   __bf16* W1lo = W1hi + W_ * W_;
    __bf16* W2hi = W1lo + W_ * W_;   __bf16* W2lo = W2hi + D_ * W_;

    init_k<<<(T_ * BD + 255) / 256, 256, 0, stream>>>(ts, y0, y, ys, sc, sci);
    split_bf16_k<<<(W_ * D_ + 255) / 256, 256, 0, stream>>>(W0, W0hi, W0lo, W_ * D_);
    split_bf16_k<<<(W_ * W_ + 255) / 256, 256, 0, stream>>>(W1, W1hi, W1lo, W_ * W_);
    split_bf16_k<<<(D_ * W_ + 255) / 256, 256, 0, stream>>>(W2, W2hi, W2lo, D_ * W_);

    dim3 blk(256);
    dim3 gW(W_ / 64, B_ / 128);   /* [1024 x 512] outputs, NBLK=4 -> (8,8)  */
    dim3 gD(D_ / 32, B_ / 128);   /* [1024 x 128] outputs, NBLK=2 -> (4,8)  */

    auto feval = [&](const float* in, float* out) {
        wmma_gemm_bias<4, D_, W_, true ><<<gW, blk, 0, stream>>>(in, W0hi, W0lo, b0, h1);
        wmma_gemm_bias<4, W_, W_, true ><<<gW, blk, 0, stream>>>(h1, W1hi, W1lo, b1, h2);
        wmma_gemm_bias<2, W_, D_, false><<<gD, blk, 0, stream>>>(h2, W2hi, W2lo, b2, out);
    };

    const int cg = (BD + 255) / 256;
    for (int s = 0; s < NSTEP; ++s) {
        step_begin_k<<<1, 1, 0, stream>>>(ts, sc, sci);
        feval(y, k[0]);
        combine_k<<<cg, blk, 0, stream>>>(ytmp, y, sc, k[0],k[1],k[2],k[3],k[4],k[5],
                                          A21c, 0, 0, 0, 0, 0, 1);
        feval(ytmp, k[1]);
        combine_k<<<cg, blk, 0, stream>>>(ytmp, y, sc, k[0],k[1],k[2],k[3],k[4],k[5],
                                          A31c, A32c, 0, 0, 0, 0, 2);
        feval(ytmp, k[2]);
        combine_k<<<cg, blk, 0, stream>>>(ytmp, y, sc, k[0],k[1],k[2],k[3],k[4],k[5],
                                          A41c, A42c, A43c, 0, 0, 0, 3);
        feval(ytmp, k[3]);
        combine_k<<<cg, blk, 0, stream>>>(ytmp, y, sc, k[0],k[1],k[2],k[3],k[4],k[5],
                                          A51c, A52c, A53c, A54c, 0, 0, 4);
        feval(ytmp, k[4]);
        combine_k<<<cg, blk, 0, stream>>>(ytmp, y, sc, k[0],k[1],k[2],k[3],k[4],k[5],
                                          A61c, A62c, A63c, A64c, A65c, 0, 5);
        feval(ytmp, k[5]);
        combine_k<<<cg, blk, 0, stream>>>(ynew, y, sc, k[0],k[1],k[2],k[3],k[4],k[5],
                                          B1c, B2c, B3c, B4c, B5c, B6c, 6);
        feval(ynew, k[6]);
        err_reduce_k<<<256, blk, 0, stream>>>(k[0],k[1],k[2],k[3],k[4],k[5],k[6], y, ynew, sc);
        step_end_k<<<1, 1, 0, stream>>>(sc, sci);
        update_k<<<cg, blk, 0, stream>>>(y, ynew, ys, sci);
    }
}
